// MultiHeadLayer_70282844832198
// MI455X (gfx1250) — matrix operand order repaired and checked by reading, not compiled, not run on hardware
//
#include <hip/hip_runtime.h>
#include <hip/hip_bf16.h>
#include <stdint.h>

typedef __attribute__((ext_vector_type(16))) _Float16 v16h;
typedef __attribute__((ext_vector_type(8)))  float    v8f;

#define B_   2
#define S_   2048
#define D_   1024
#define H_   16
#define HD   64
#define SEQLEN 1024
#define NEG_BIG (-1.0e30f)

// ---- Tensor Data Mover availability (device pass only; host pass -> 0) -----
#if __has_builtin(__builtin_amdgcn_tensor_load_to_lds)
#define USE_TDM 1
#if __has_include(<hip/amd_detail/amd_gfx1250_TDM.h>)
#define TDM_6ARG 1  // amdgpu-toolchain / therock headers -> 6-arg builtin
#endif
#endif

// ---- WMMA fragment index maps (CDNA5 ISA §7.12.2, 16-bit data) --
__device__ __forceinline__ int kmapA(int v, int g) {
  return (v < 4) ? (8 * g + 2 * v) : (16 + 8 * g + 2 * (v - 4));
}
__device__ __forceinline__ int kmapB(int v, int g) { return kmapA(v, g); }

__device__ __forceinline__ v8f wmma_f16(v16h a, v16h b, v8f c) {
  return __builtin_amdgcn_wmma_f32_16x16x32_f16(false, a, false, b, (short)0, c,
                                                false, false);
}

// A fragment from row-major f16 matrix; p -> (row0, col0), leading dim ld.
__device__ __forceinline__ v16h load_a_f16(const _Float16* p, int ld) {
  const int lane = threadIdx.x & 31, m = lane & 15, g = lane >> 4;
  const _Float16* row = p + (size_t)m * ld;
  v16h a;
#pragma unroll
  for (int v = 0; v < 8; ++v) {
    const int k = kmapA(v, g);
    a[2 * v]     = row[k];
    a[2 * v + 1] = row[k + 1];
  }
  return a;
}

// A fragment from row-major f32 matrix (converted to f16 on the fly).
__device__ __forceinline__ v16h load_a_f32(const float* p, int ld) {
  const int lane = threadIdx.x & 31, m = lane & 15, g = lane >> 4;
  const float* row = p + (size_t)m * ld;
  v16h a;
#pragma unroll
  for (int v = 0; v < 8; ++v) {
    const int k = kmapA(v, g);
    a[2 * v]     = (_Float16)row[k];
    a[2 * v + 1] = (_Float16)row[k + 1];
  }
  return a;
}

// B fragment where B[k][n] = src[n][k] (src row-major, ld). p -> (n0, k0).
__device__ __forceinline__ v16h load_bT_f16(const _Float16* p, int ld) {
  const int lane = threadIdx.x & 31, n = lane & 15, g = lane >> 4;
  const _Float16* row = p + (size_t)n * ld;
  v16h b;
#pragma unroll
  for (int v = 0; v < 8; ++v) {
    const int k = kmapB(v, g);
    b[2 * v]     = row[k];
    b[2 * v + 1] = row[k + 1];
  }
  return b;
}

// B fragment from row-major f32 src: B[k][n] = src[k][n]; p -> (k0, n0).
__device__ __forceinline__ v16h load_b_f32(const float* p, int ld) {
  const int lane = threadIdx.x & 31, n = lane & 15, g = lane >> 4;
  v16h b;
#pragma unroll
  for (int v = 0; v < 8; ++v) {
    const int k = kmapB(v, g);
    b[2 * v]     = (_Float16)p[(size_t)k * ld + n];
    b[2 * v + 1] = (_Float16)p[(size_t)(k + 1) * ld + n];
  }
  return b;
}

#ifdef USE_TDM
typedef unsigned int uint4v __attribute__((ext_vector_type(4)));
typedef int          int8v  __attribute__((ext_vector_type(8)));
typedef int          int4v  __attribute__((ext_vector_type(4)));

// Issue a TDM load of a 1D/2D f16 tile into LDS (CDNA5 ISA).
// tile_d1 == 0 -> 1D copy of tile_d0 elements. Tracked on TENSORcnt.
__device__ __forceinline__ void tdm_load_tile(unsigned lds_byte_addr,
                                              const _Float16* gsrc,
                                              unsigned tile_d0, unsigned tile_d1,
                                              unsigned tens_d0, unsigned tens_d1,
                                              unsigned stride0_elems) {
  const unsigned long long ga = (unsigned long long)(uintptr_t)gsrc;
  uint4v g0;
  g0[0] = 1u;                                    // count=1, user descriptor
  g0[1] = lds_byte_addr;                         // lds_addr
  g0[2] = (unsigned)ga;                          // global_addr[31:0]
  g0[3] = (unsigned)((ga >> 32) & 0x01FFFFFFu)   // global_addr[56:32]
          | (2u << 30);                          // type = 2 ("image")
  int8v g1;
  g1[0] = (int)(1u << 16);                       // data_size = 1 -> 2 bytes
  g1[1] = (int)((tens_d0 & 0xFFFFu) << 16);                                // dim0[15:0] @63:48
  g1[2] = (int)(((tens_d0 >> 16) & 0xFFFFu) | ((tens_d1 & 0xFFFFu) << 16));// dim0 hi | dim1 lo
  g1[3] = (int)(((tens_d1 >> 16) & 0xFFFFu) | ((tile_d0 & 0xFFFFu) << 16));// dim1 hi | tile0
  g1[4] = (int)(tile_d1 & 0xFFFFu);              // tile_dim1 (tile_dim2 = 0)
  g1[5] = (int)stride0_elems;                    // tensor_dim0_stride[31:0]
  g1[6] = 0;
  g1[7] = 0;
  const int4v z4 = {0, 0, 0, 0};
#ifdef TDM_6ARG
  const int8v z8 = {0, 0, 0, 0, 0, 0, 0, 0};
  __builtin_amdgcn_tensor_load_to_lds(g0, g1, z4, z4, z8, 0);
#else
  __builtin_amdgcn_tensor_load_to_lds(g0, g1, z4, z4, 0);
#endif
}
#endif  // USE_TDM

// ---------------- Kernel 1: per-head QKV projection (x f32 -> q/k/v f16) ----
// grid (S/16, H, B*3). One wave computes a 16x64 tile: 8 WMMAs.
// q,k stored [B,H,S,d]; v stored transposed [B,H,d,S] for contiguous B-frags.
__global__ void __launch_bounds__(32) qkv_proj_kernel(
    const float* __restrict__ x,
    const float* __restrict__ Wq, const float* __restrict__ bq,
    const float* __restrict__ Wk, const float* __restrict__ bk,
    const float* __restrict__ Wv, const float* __restrict__ bv,
    _Float16* __restrict__ qf, _Float16* __restrict__ kf,
    _Float16* __restrict__ vT) {
  const int s0  = blockIdx.x * 16;
  const int h   = blockIdx.y;
  const int b   = blockIdx.z / 3;
  const int sel = blockIdx.z % 3;
  const int lane = threadIdx.x & 31, nn = lane & 15, g = lane >> 4;

  const float* W    = ((sel == 0) ? Wq : (sel == 1) ? Wk : Wv) + (size_t)h * HD * HD;
  const float* bias = ((sel == 0) ? bq : (sel == 1) ? bk : bv) + (size_t)h * HD;

  const float* xa = x + ((size_t)b * S_ + s0) * D_ + h * HD;
  const v16h a0 = load_a_f32(xa, D_);
  const v16h a1 = load_a_f32(xa + 32, D_);

#pragma unroll
  for (int n0 = 0; n0 < HD; n0 += 16) {
    v8f c;
    const float bias_n = bias[n0 + nn];
#pragma unroll
    for (int r = 0; r < 8; ++r) c[r] = bias_n;
    c = wmma_f16(a0, load_b_f32(W + n0, HD), c);
    c = wmma_f16(a1, load_b_f32(W + (size_t)32 * HD + n0, HD), c);

    if (sel < 2) {
      _Float16* out = (sel == 0 ? qf : kf) + ((size_t)b * H_ + h) * S_ * HD;
#pragma unroll
      for (int r = 0; r < 8; ++r)
        out[(size_t)(s0 + r + 8 * g) * HD + n0 + nn] = (_Float16)c[r];
    } else {
      _Float16* out = vT + ((size_t)b * H_ + h) * HD * S_;
#pragma unroll
      for (int r = 0; r < 8; ++r)
        out[(size_t)(n0 + nn) * S_ + s0 + r + 8 * g] = (_Float16)c[r];
    }
  }
}

// ---------------- Kernel 2: Wo -> WoT (f16, transposed [N][K]) --------------
__global__ void __launch_bounds__(256) wo_transpose_kernel(
    const float* __restrict__ Wo, _Float16* __restrict__ WoT) {
  const int idx = blockIdx.x * 256 + threadIdx.x;  // over D*D
  const int n = idx / D_, k = idx % D_;
  WoT[idx] = (_Float16)Wo[(size_t)k * D_ + n];
}

// ---------------- Kernel 3: flash attention ---------------------------------
// grid (S/64, H, B), 128 threads = 4 waves. Each wave owns one 16-query tile;
// the 4 waves share K/V tiles staged in LDS by the Tensor Data Mover
// (double-buffered, s_wait_tensorcnt), falling back to cooperative b128 loads.
__global__ void __launch_bounds__(128) attn_kernel(
    const _Float16* __restrict__ qf, const _Float16* __restrict__ kf,
    const _Float16* __restrict__ vT, _Float16* __restrict__ o) {
  __shared__ _Float16 kbuf[2][32 * HD];   // keys x features (row-major)
  __shared__ _Float16 vbuf[2][HD * 32];   // features x keys (row-major)
  __shared__ _Float16 pbuf[4][16 * 32];   // per-wave P staging

  const int w    = threadIdx.x >> 5;      // wave id 0..3
  const int lane = threadIdx.x & 31, nn = lane & 15, g = lane >> 4;
  const int s0 = blockIdx.x * 64 + w * 16;
  const int h  = blockIdx.y;
  const int b  = blockIdx.z;

  const _Float16* qh = qf + ((size_t)b * H_ + h) * S_ * HD;
  const _Float16* kh = kf + ((size_t)b * H_ + h) * S_ * HD;
  const _Float16* vh = vT + ((size_t)b * H_ + h) * HD * S_;

  const v16h aq0 = load_a_f16(qh + (size_t)s0 * HD, HD);
  const v16h aq1 = load_a_f16(qh + (size_t)s0 * HD + 32, HD);

  v8f acc[4];
  float mrow[8], lrow[8];
#pragma unroll
  for (int r = 0; r < 8; ++r) {
    mrow[r] = NEG_BIG;
    lrow[r] = 0.0f;
  }
#pragma unroll
  for (int f = 0; f < 4; ++f)
#pragma unroll
    for (int r = 0; r < 8; ++r) acc[f][r] = 0.0f;

  const bool maskable = (s0 == S_ - 16);  // wave holding query row S-1

  // Stage K tile (32 keys x 64 feats, contiguous) and V^T tile (64 feats x
  // 32 keys, row stride S) for keys [t0, t0+32). Each wave moves a quarter.
  auto issue_tiles = [&](int t0, int buf) {
#ifdef USE_TDM
    tdm_load_tile((unsigned)(uintptr_t)&kbuf[buf][(size_t)w * 8 * HD],
                  kh + (size_t)(t0 + 8 * w) * HD,
                  /*tile_d0=*/8 * HD, /*tile_d1=*/0,        // 1D, 512 elems
                  /*tens_d0=*/S_ * HD, /*tens_d1=*/1, /*stride0=*/S_ * HD);
    tdm_load_tile((unsigned)(uintptr_t)&vbuf[buf][(size_t)w * 16 * 32],
                  vh + (size_t)(w * 16) * S_ + t0,
                  /*tile_d0=*/32, /*tile_d1=*/16,           // 2D 16x32
                  /*tens_d0=*/S_, /*tens_d1=*/HD, /*stride0=*/S_);
#else
    const unsigned tid = threadIdx.x;
    const uint4* ksrc = (const uint4*)(kh + (size_t)t0 * HD);
    uint4* kdst = (uint4*)&kbuf[buf][0];
    kdst[tid]       = ksrc[tid];
    kdst[tid + 128] = ksrc[tid + 128];
    const int rr = tid >> 1, hv = tid & 1;
    const uint4* vsrc = (const uint4*)(vh + (size_t)rr * S_ + t0 + hv * 16);
    uint4* vdst = (uint4*)&vbuf[buf][rr * 32 + hv * 16];
    vdst[0] = vsrc[0];
    vdst[1] = vsrc[1];
#endif
  };

#ifdef USE_TDM
  issue_tiles(0, 0);
#endif

  for (int t0 = 0; t0 < S_; t0 += 32) {
#ifdef USE_TDM
    const int cur = (t0 >> 5) & 1;
    __syncthreads();  // all waves done with the buffer we are about to refill
    if (t0 + 32 < S_) {
      issue_tiles(t0 + 32, cur ^ 1);
      __builtin_amdgcn_s_wait_tensorcnt(2);  // in-order: current 2 loads done
    } else {
      __builtin_amdgcn_s_wait_tensorcnt(0);
    }
    __syncthreads();
#else
    const int cur = 0;
    __syncthreads();
    issue_tiles(t0, 0);
    __syncthreads();
#endif
    const _Float16* kt = &kbuf[cur][0];
    const _Float16* vt = &vbuf[cur][0];

    // ---- scores: two 16x16 tiles over keys [t0, t0+32) (from LDS)
    v8f c0, c1;
#pragma unroll
    for (int r = 0; r < 8; ++r) { c0[r] = 0.0f; c1[r] = 0.0f; }
    c0 = wmma_f16(aq0, load_bT_f16(kt, HD), c0);
    c0 = wmma_f16(aq1, load_bT_f16(kt + 32, HD), c0);
    c1 = wmma_f16(aq0, load_bT_f16(kt + 16 * HD, HD), c1);
    c1 = wmma_f16(aq1, load_bT_f16(kt + 16 * HD + 32, HD), c1);

    // ---- mask: query row S-1, keys >= S-SEQLEN -> large negative
    if (maskable) {
#pragma unroll
      for (int r = 0; r < 8; ++r) {
        const bool lastrow = (r + 8 * g) == 15;
        c0[r] = (lastrow && (t0 + nn)      >= (S_ - SEQLEN)) ? NEG_BIG : c0[r];
        c1[r] = (lastrow && (t0 + 16 + nn) >= (S_ - SEQLEN)) ? NEG_BIG : c1[r];
      }
    }

    // ---- online softmax (row = 16 lanes of the same half-wave)
    float p0[8], p1[8];
#pragma unroll
    for (int r = 0; r < 8; ++r) {
      float mx = fmaxf(c0[r], c1[r]);
#pragma unroll
      for (int off = 1; off < 16; off <<= 1) mx = fmaxf(mx, __shfl_xor(mx, off));
      const float newm  = fmaxf(mrow[r], mx);
      const float alpha = __expf(mrow[r] - newm);
      p0[r] = __expf(c0[r] - newm);
      p1[r] = __expf(c1[r] - newm);
      float rs = p0[r] + p1[r];
#pragma unroll
      for (int off = 1; off < 16; off <<= 1) rs += __shfl_xor(rs, off);
      lrow[r] = lrow[r] * alpha + rs;
      mrow[r] = newm;
#pragma unroll
      for (int f = 0; f < 4; ++f) acc[f][r] *= alpha;
      pbuf[w][(r + 8 * g) * 32 + nn]      = (_Float16)p0[r];
      pbuf[w][(r + 8 * g) * 32 + 16 + nn] = (_Float16)p1[r];
    }

    const v16h ap = load_a_f16(&pbuf[w][0], 32);  // C-frag -> A-frag via LDS
#pragma unroll
    for (int f = 0; f < 4; ++f)
      acc[f] = wmma_f16(ap, load_bT_f16(vt + (size_t)(f * 16) * 32, 32), acc[f]);
  }

  // ---- epilogue: divide by rowsum and post-softmax SCALE (= sqrt(D) = 32)
  _Float16* ob = o + ((size_t)b * S_ + s0) * D_ + h * HD;
#pragma unroll
  for (int r = 0; r < 8; ++r) {
    const float inv = 1.0f / (lrow[r] * 32.0f);
#pragma unroll
    for (int f = 0; f < 4; ++f)
      ob[(size_t)(r + 8 * g) * D_ + f * 16 + nn] = (_Float16)(acc[f][r] * inv);
  }
}

// ---------------- Kernel 4: output projection + residual --------------------
// grid (S/16, D/64, B). One wave: 16x64 tile of y = x + o @ Wo + bo. 128 WMMAs.
__global__ void __launch_bounds__(32) oproj_kernel(
    const _Float16* __restrict__ o, const _Float16* __restrict__ WoT,
    const float* __restrict__ bo, const float* __restrict__ x,
    float* __restrict__ y) {
  const int s0 = blockIdx.x * 16;
  const int n0 = blockIdx.y * 64;
  const int b  = blockIdx.z;
  const int lane = threadIdx.x & 31, nn = lane & 15, g = lane >> 4;

  const _Float16* orow = o + ((size_t)b * S_ + s0) * D_;

  v8f c[4];
#pragma unroll
  for (int f = 0; f < 4; ++f) {
    const float bias_n = bo[n0 + f * 16 + nn];
#pragma unroll
    for (int r = 0; r < 8; ++r) c[f][r] = bias_n;
  }

  for (int k0 = 0; k0 < D_; k0 += 32) {
    if (k0 + 32 < D_) __builtin_prefetch(orow + k0 + 32, 0, 0);
    const v16h a = load_a_f16(orow + k0, D_);
#pragma unroll
    for (int f = 0; f < 4; ++f)
      c[f] = wmma_f16(a, load_bT_f16(WoT + (size_t)(n0 + f * 16) * D_ + k0, D_),
                      c[f]);
  }

  const float* xb = x + ((size_t)b * S_ + s0) * D_;
  float* yb = y + ((size_t)b * S_ + s0) * D_;
#pragma unroll
  for (int f = 0; f < 4; ++f)
#pragma unroll
    for (int r = 0; r < 8; ++r) {
      const size_t off = (size_t)(r + 8 * g) * D_ + n0 + f * 16 + nn;
      yb[off] = xb[off] + c[f][r];
    }
}

// ---------------- Kernel 5: LayerNorm ---------------------------------------
__global__ void __launch_bounds__(256) ln_kernel(
    const float* __restrict__ y, const float* __restrict__ gam,
    const float* __restrict__ bet, float* __restrict__ out) {
  const int row = blockIdx.x;
  const int tid = threadIdx.x;
  const float* yr = y + (size_t)row * D_;

  float v[4];
  float s = 0.0f, s2 = 0.0f;
#pragma unroll
  for (int i = 0; i < 4; ++i) {
    v[i] = yr[tid + 256 * i];
    s += v[i];
    s2 += v[i] * v[i];
  }
#pragma unroll
  for (int off = 16; off >= 1; off >>= 1) {
    s  += __shfl_xor(s, off);
    s2 += __shfl_xor(s2, off);
  }
  __shared__ float ws1[8], ws2[8];
  const int w = tid >> 5;
  if ((tid & 31) == 0) { ws1[w] = s; ws2[w] = s2; }
  __syncthreads();
  float ts = 0.0f, ts2 = 0.0f;
#pragma unroll
  for (int j = 0; j < 8; ++j) { ts += ws1[j]; ts2 += ws2[j]; }
  const float mu   = ts * (1.0f / D_);
  const float var  = ts2 * (1.0f / D_) - mu * mu;
  const float rstd = rsqrtf(var + 1e-5f);
#pragma unroll
  for (int i = 0; i < 4; ++i) {
    const int cidx = tid + 256 * i;
    out[(size_t)row * D_ + cidx] = (v[i] - mu) * rstd * gam[cidx] + bet[cidx];
  }
}

extern "C" void kernel_launch(void* const* d_in, const int* in_sizes, int n_in,
                              void* d_out, int out_size, void* d_ws,
                              size_t ws_size, hipStream_t stream) {
  (void)in_sizes; (void)n_in; (void)out_size; (void)ws_size;
  const float* x    = (const float*)d_in[0];
  const float* Wq   = (const float*)d_in[1];
  const float* bq   = (const float*)d_in[2];
  const float* Wk   = (const float*)d_in[3];
  const float* bk   = (const float*)d_in[4];
  const float* Wv   = (const float*)d_in[5];
  const float* bv   = (const float*)d_in[6];
  const float* Wo   = (const float*)d_in[7];
  const float* bo   = (const float*)d_in[8];
  const float* ln_g = (const float*)d_in[9];
  const float* ln_b = (const float*)d_in[10];
  float* out = (float*)d_out;

  char* ws = (char*)d_ws;
  _Float16* qf  = (_Float16*)(ws);                          // 8 MB  [B,H,S,d]
  _Float16* kf  = (_Float16*)(ws + ((size_t)8  << 20));     // 8 MB  [B,H,S,d]
  _Float16* vT  = (_Float16*)(ws + ((size_t)16 << 20));     // 8 MB  [B,H,d,S]
  _Float16* of  = (_Float16*)(ws + ((size_t)24 << 20));     // 8 MB  [B,S,D]
  float*    yf  = (float*)   (ws + ((size_t)32 << 20));     // 16 MB [B,S,D]
  _Float16* WoT = (_Float16*)(ws + ((size_t)48 << 20));     // 2 MB  [D,D] T

  qkv_proj_kernel<<<dim3(S_ / 16, H_, B_ * 3), 32, 0, stream>>>(
      x, Wq, bq, Wk, bk, Wv, bv, qf, kf, vT);
  wo_transpose_kernel<<<dim3(D_ * D_ / 256), 256, 0, stream>>>(Wo, WoT);
  attn_kernel<<<dim3(S_ / 64, H_, B_), 128, 0, stream>>>(qf, kf, vT, of);
  oproj_kernel<<<dim3(S_ / 16, D_ / 64, B_), 32, 0, stream>>>(of, WoT, bo, x, yf);
  ln_kernel<<<dim3(B_ * S_), 256, 0, stream>>>(yf, ln_g, ln_b, out);
}
